// ElasticCosFace_16200616640758
// MI455X (gfx1250) — matrix operand order repaired and checked by reading, not compiled, not run on hardware
//
#include <hip/hip_runtime.h>
#include <hip/hip_bf16.h>

#define B_SZ 256
#define D_SZ 512
#define C_SZ 100000
#define S_SCALE 64.0f
#define M_MEAN 0.35f
#define M_STD 0.0125f
#define EPS_CLIP 1e-7f

typedef __attribute__((ext_vector_type(16))) __bf16 v16bf;
typedef __attribute__((ext_vector_type(8)))  float  v8f;

union FragAB {
  v16bf v;
  uint4 q[2];
};

__device__ __forceinline__ unsigned short f32_to_bf16_rne(float f) {
  unsigned int u = __float_as_uint(f);
  u += 0x7FFFu + ((u >> 16) & 1u);   // round-to-nearest-even on bit 16
  return (unsigned short)(u >> 16);
}
__device__ __forceinline__ float bf16_to_f32(unsigned short h) {
  return __uint_as_float(((unsigned int)h) << 16);
}

// ---------------------------------------------------------------------------
// Kernel 1: per-row feature norms, inverse norms, elastic margin, bf16 hi/lo
// split of feats (normalization folded into the GEMM epilogue).
// ---------------------------------------------------------------------------
__global__ void __launch_bounds__(128)
rowprep_kernel(const float* __restrict__ feats,
               const float* __restrict__ margin_raw,
               float* __restrict__ norms_out,
               float* __restrict__ invA,
               float* __restrict__ marg,
               unsigned short* __restrict__ Ahi,
               unsigned short* __restrict__ Alo) {
  const int b = blockIdx.x;
  const int tid = threadIdx.x;
  float s = 0.f;
  for (int d = tid; d < D_SZ; d += 128) {
    float f = feats[b * D_SZ + d];
    s += f * f;
  }
  __shared__ float red[128];
  red[tid] = s;
  __syncthreads();
  for (int off = 64; off > 0; off >>= 1) {
    if (tid < off) red[tid] += red[tid + off];
    __syncthreads();
  }
  if (tid == 0) {
    float nrm = sqrtf(red[0]);
    norms_out[b] = nrm;
    invA[b] = 1.0f / fmaxf(nrm, 1e-12f);
    float m = M_MEAN + M_STD * margin_raw[b];
    marg[b] = fminf(fmaxf(m, M_MEAN - M_STD), M_MEAN + M_STD);
  }
  for (int d = tid; d < D_SZ; d += 128) {
    float f = feats[b * D_SZ + d];
    unsigned short hi = f32_to_bf16_rne(f);
    Ahi[b * D_SZ + d] = hi;
    Alo[b * D_SZ + d] = f32_to_bf16_rne(f - bf16_to_f32(hi));
  }
}

// ---------------------------------------------------------------------------
// Kernel 2: per-column kernel norms (coalesced stride-C reads; warms L2 with
// the weight matrix, which we then keep resident by writing outputs NT).
// ---------------------------------------------------------------------------
__global__ void __launch_bounds__(256)
colnorm_kernel(const float* __restrict__ Wk, float* __restrict__ invW) {
  const int c = blockIdx.x * 256 + threadIdx.x;
  if (c >= C_SZ) return;
  float s = 0.f;
  for (int d = 0; d < D_SZ; ++d) {
    float w = Wk[(size_t)d * C_SZ + c];
    s += w * w;
  }
  invW[c] = 1.0f / fmaxf(sqrtf(s), 1e-12f);
}

// ---------------------------------------------------------------------------
// Kernel 3: split-precision bf16 WMMA GEMM + fused epilogue.
// Block: 256 threads (8 waves), computes 16 columns x all 256 rows
// (2 row-strips of 16 per wave). K-loop: 16 steps of K=32, 3 WMMAs per
// accumulator per step (hi*hi + hi*lo + lo*hi). LDS B-tile double buffered:
// one barrier per step, staging overlaps WMMA.
// ---------------------------------------------------------------------------
__device__ __forceinline__ void stage_elem(const float* __restrict__ Wk,
                                           int k0, int c0, int i,
                                           unsigned short (&hi)[16][40],
                                           unsigned short (&lo)[16][40]) {
  const int kk = i >> 4;
  const int nn = i & 15;
  float w = Wk[(size_t)(k0 + kk) * C_SZ + (c0 + nn)];
  unsigned short h = f32_to_bf16_rne(w);
  hi[nn][kk] = h;
  lo[nn][kk] = f32_to_bf16_rne(w - bf16_to_f32(h));
}

__device__ __forceinline__ void epilogue_store(float accv, int row, int col, float iw,
                                               const float* __restrict__ invA,
                                               const float* __restrict__ marg,
                                               const int* __restrict__ labels,
                                               float* __restrict__ out_pre,
                                               float* __restrict__ out_log,
                                               float* __restrict__ out_oh) {
  float cosv = accv * invA[row] * iw;
  cosv = fminf(fmaxf(cosv, -1.0f + EPS_CLIP), 1.0f - EPS_CLIP);
  float oh = (labels[row] == col) ? 1.0f : 0.0f;
  size_t idx = (size_t)row * C_SZ + col;
  // Streamed, write-once outputs: non-temporal so they don't evict the
  // L2-resident weight matrix.
  __builtin_nontemporal_store(cosv * S_SCALE, &out_pre[idx]);
  __builtin_nontemporal_store((cosv - oh * marg[row]) * S_SCALE, &out_log[idx]);
  __builtin_nontemporal_store(oh, &out_oh[idx]);
}

__global__ void __launch_bounds__(256)
gemm_kernel(const float* __restrict__ Wk,
            const unsigned short* __restrict__ Ahi,
            const unsigned short* __restrict__ Alo,
            const float* __restrict__ invA,
            const float* __restrict__ invW,
            const float* __restrict__ marg,
            const int* __restrict__ labels,
            float* __restrict__ out_pre,
            float* __restrict__ out_log,
            float* __restrict__ out_oh) {
  // Double-buffered LDS B tile: [buf][n][k], rows padded to 40 shorts (80B)
  // so the 16B ds_load_b128 fragment reads stay aligned.
  __shared__ __align__(16) unsigned short sBhi[2][16][40];
  __shared__ __align__(16) unsigned short sBlo[2][16][40];

  const int tid  = threadIdx.x;
  const int lane = tid & 31;
  const int wave = tid >> 5;
  const int half = lane >> 4;     // selects K half of the fragment
  const int nloc = lane & 15;     // A row within strip / B column / D column
  const int c0 = blockIdx.x * 16;
  const int m0 = wave * 16;       // row strip 0
  const int m1 = m0 + 128;        // row strip 1

  v8f acc0 = {0.f, 0.f, 0.f, 0.f, 0.f, 0.f, 0.f, 0.f};
  v8f acc1 = {0.f, 0.f, 0.f, 0.f, 0.f, 0.f, 0.f, 0.f};

  const unsigned short* aHi0 = Ahi + (size_t)(m0 + nloc) * D_SZ;
  const unsigned short* aLo0 = Alo + (size_t)(m0 + nloc) * D_SZ;
  const unsigned short* aHi1 = Ahi + (size_t)(m1 + nloc) * D_SZ;
  const unsigned short* aLo1 = Alo + (size_t)(m1 + nloc) * D_SZ;

  // Prologue: stage first tile into buffer 0 (2 elements/thread, unrolled).
  stage_elem(Wk, 0, c0, tid,       sBhi[0], sBlo[0]);
  stage_elem(Wk, 0, c0, tid + 256, sBhi[0], sBlo[0]);

  for (int ks = 0; ks < D_SZ / 32; ++ks) {
    const int k0 = ks * 32;
    const int cur = ks & 1;
    __syncthreads();  // staging of buf[cur] visible; buf[cur^1] reads retired

    // Stage next tile into the other buffer (overlaps the WMMAs below).
    if (ks + 1 < D_SZ / 32) {
      stage_elem(Wk, k0 + 32, c0, tid,       sBhi[cur ^ 1], sBlo[cur ^ 1]);
      stage_elem(Wk, k0 + 32, c0, tid + 256, sBhi[cur ^ 1], sBlo[cur ^ 1]);
    }

    // B fragment: VGPRs0-3 = K in [half*8, +8), VGPRs4-7 = K in [16+half*8, +8) for column nloc.
    FragAB bh, bl;
    bh.q[0] = *(const uint4*)&sBhi[cur][nloc][half * 8];
    bh.q[1] = *(const uint4*)&sBhi[cur][nloc][half * 8 + 16];
    bl.q[0] = *(const uint4*)&sBlo[cur][nloc][half * 8];
    bl.q[1] = *(const uint4*)&sBlo[cur][nloc][half * 8 + 16];

    // A fragment: VGPRs0-3 = K in [k0+half*8, +8), VGPRs4-7 = K in [k0+16+half*8, +8).
    const int ka = k0 + half * 8;
    FragAB ah0, al0, ah1, al1;
    ah0.q[0] = *(const uint4*)(aHi0 + ka);
    ah0.q[1] = *(const uint4*)(aHi0 + ka + 16);
    al0.q[0] = *(const uint4*)(aLo0 + ka);
    al0.q[1] = *(const uint4*)(aLo0 + ka + 16);
    ah1.q[0] = *(const uint4*)(aHi1 + ka);
    ah1.q[1] = *(const uint4*)(aHi1 + ka + 16);
    al1.q[0] = *(const uint4*)(aLo1 + ka);
    al1.q[1] = *(const uint4*)(aLo1 + ka + 16);

    // Split-precision product: hi*hi + hi*lo + lo*hi (lo*lo ~2^-18, dropped).
    acc0 = __builtin_amdgcn_wmma_f32_16x16x32_bf16(false, ah0.v, false, bh.v, (short)0, acc0, false, false);
    acc0 = __builtin_amdgcn_wmma_f32_16x16x32_bf16(false, ah0.v, false, bl.v, (short)0, acc0, false, false);
    acc0 = __builtin_amdgcn_wmma_f32_16x16x32_bf16(false, al0.v, false, bh.v, (short)0, acc0, false, false);
    acc1 = __builtin_amdgcn_wmma_f32_16x16x32_bf16(false, ah1.v, false, bh.v, (short)0, acc1, false, false);
    acc1 = __builtin_amdgcn_wmma_f32_16x16x32_bf16(false, ah1.v, false, bl.v, (short)0, acc1, false, false);
    acc1 = __builtin_amdgcn_wmma_f32_16x16x32_bf16(false, al1.v, false, bh.v, (short)0, acc1, false, false);
  }

  // Epilogue. C/D layout: lanes 0-15 -> N=lane, M=r; lanes 16-31 -> N=lane-16, M=r+8.
  const int col = c0 + nloc;
  const float iw = invW[col];
  const int mb = half * 8;
#pragma unroll
  for (int r = 0; r < 8; ++r) {
    epilogue_store(acc0[r], m0 + mb + r, col, iw, invA, marg, labels, out_pre, out_log, out_oh);
    epilogue_store(acc1[r], m1 + mb + r, col, iw, invA, marg, labels, out_pre, out_log, out_oh);
  }
}

// ---------------------------------------------------------------------------
extern "C" void kernel_launch(void* const* d_in, const int* in_sizes, int n_in,
                              void* d_out, int out_size, void* d_ws, size_t ws_size,
                              hipStream_t stream) {
  (void)in_sizes; (void)n_in; (void)out_size; (void)ws_size;

  const float* feats      = (const float*)d_in[0];
  const float* kernelW    = (const float*)d_in[1];
  const float* margin_raw = (const float*)d_in[2];
  const int*   labels     = (const int*)d_in[3];

  float* out      = (float*)d_out;
  float* out_pre  = out;                                  // [B,C]
  float* out_log  = out + (size_t)B_SZ * C_SZ;            // [B,C]
  float* norms    = out + 2 * (size_t)B_SZ * C_SZ;        // [B,1]
  float* out_oh   = norms + B_SZ;                         // [B,C]

  char* w = (char*)d_ws;
  unsigned short* Ahi = (unsigned short*)w;  w += (size_t)B_SZ * D_SZ * sizeof(unsigned short);
  unsigned short* Alo = (unsigned short*)w;  w += (size_t)B_SZ * D_SZ * sizeof(unsigned short);
  float* invA = (float*)w;  w += (size_t)B_SZ * sizeof(float);
  float* marg = (float*)w;  w += (size_t)B_SZ * sizeof(float);
  float* invW = (float*)w;  w += (size_t)C_SZ * sizeof(float);

  rowprep_kernel<<<B_SZ, 128, 0, stream>>>(feats, margin_raw, norms, invA, marg, Ahi, Alo);
  colnorm_kernel<<<(C_SZ + 255) / 256, 256, 0, stream>>>(kernelW, invW);
  gemm_kernel<<<C_SZ / 16, 256, 0, stream>>>(kernelW, Ahi, Alo, invA, invW, marg, labels,
                                             out_pre, out_log, out_oh);
}
